// Absolute_attention_74363063763520
// MI455X (gfx1250) — compile-verified, matrix operand order checked by reading, not run on hardware
//
#include <hip/hip_runtime.h>

// ---------------------------------------------------------------- types
typedef __attribute__((ext_vector_type(16))) _Float16 v16h;
typedef __attribute__((ext_vector_type(8)))  _Float16 v8h;
typedef __attribute__((ext_vector_type(8)))  float    v8f;
typedef __attribute__((ext_vector_type(4)))  int      v4i;

#define GLOBAL_AS __attribute__((address_space(1)))
#define LDS_AS    __attribute__((address_space(3)))

#define B_   2
#define L_   2048
#define DIM_ 512
#define H_   8
#define HD_  64
#define TD2_ 128      // 2*TD time features
#define LIN_ 1536
#define BL_  (B_*L_)  // 4096

#if __has_builtin(__builtin_amdgcn_global_load_async_to_lds_b128)
#define HAS_ASYNC_LDS 1
#else
#define HAS_ASYNC_LDS 0
#endif

// ---------------------------------------------------------------- f32 -> f16 convert
__global__ void cvt_f32_f16(const float* __restrict__ src, _Float16* __restrict__ dst, long n) {
    long i = (long)blockIdx.x * blockDim.x + threadIdx.x;
    if (i < n) dst[i] = (_Float16)src[i];
}

// bias broadcast-add over last dim (DIM_)
__global__ void addbias_kernel(float* __restrict__ x, const float* __restrict__ bias, long n) {
    long i = (long)blockIdx.x * blockDim.x + threadIdx.x;
    if (i < n) x[i] += bias[i & (DIM_ - 1)];
}

// ---------------------------------------------------------------- time features
// time[l,h,f] = (cos±sin)((l+delta[h])*angle[h,f&63]) / 8
// writes row-major dup [b][h][L][128] and transposed dup [b][h][128][L] (f16)
__global__ void time_kernel(const float* __restrict__ angle, const float* __restrict__ delta,
                            _Float16* __restrict__ trm, _Float16* __restrict__ ttr) {
    long i = (long)blockIdx.x * blockDim.x + threadIdx.x;
    if (i >= (long)H_ * L_ * TD2_) return;
    int f = (int)(i & (TD2_ - 1));
    int l = (int)((i >> 7) & (L_ - 1));
    int h = (int)(i >> 7) / L_;
    int fa = f & (HD_ - 1);
    float ang = ((float)l + delta[h]) * angle[h * HD_ + fa];
    float s, c;
    sincosf(ang, &s, &c);
    float val = ((f < HD_) ? (c + s) : (c - s)) * 0.125f;
    _Float16 hv = (_Float16)val;
    #pragma unroll
    for (int b = 0; b < B_; ++b) {
        long bh = (long)(b * H_ + h);
        trm[(bh * L_ + l) * TD2_ + f] = hv;
        ttr[(bh * TD2_ + f) * L_ + l] = hv;
    }
}

// ---------------------------------------------------------------- pack B into WMMA fragment order
// dst[((nt*Kt+kt)*32+lane)*16+j] = (f16) src[k*rowStride + n],  n = nt*16+(lane&15)
// K-map: j<8 -> kt*32 + koff + j ; j>=8 -> kt*32 + 16 + koff + (j-8), koff = (lane&16)?8:0
__global__ void pack_b_f32(const float* __restrict__ src, _Float16* __restrict__ dst,
                           int K, int N, long rowStride, long srcBatchStride, long dstBatchStride) {
    int batch = blockIdx.y;
    src += (long)batch * srcBatchStride;
    dst += (long)batch * dstBatchStride;
    int Kt = K >> 5, Nt = N >> 4;
    int idx = blockIdx.x * blockDim.x + threadIdx.x;
    if (idx >= Nt * Kt * 32) return;
    int lane = idx & 31;
    int kt = (idx >> 5) % Kt;
    int nt = (idx >> 5) / Kt;
    int koff = (lane & 16) ? 8 : 0;
    int n = nt * 16 + (lane & 15);
    _Float16* out = dst + (long)idx * 16;
    #pragma unroll
    for (int j = 0; j < 16; ++j) {
        int k = kt * 32 + (j < 8 ? koff + j : 16 + koff + (j - 8));
        out[j] = (_Float16)src[(long)k * rowStride + n];
    }
}

// ---------------------------------------------------------------- batched WMMA GEMM, LDS-staged B
// C[M,N] = A[M,K](f16 row-major, lda) x Bp(packed f16) + bias
// Block = 256 threads = 8 waves; block owns a 128(M) x 64(N) tile (wave w -> M-tile w).
// B fragments (4KB per K-step) are staged once per block into double-buffered LDS via the
// gfx1250 async global->LDS path (fallback: register staging + ds_store).
// Requires: M % 128 == 0, N % 64 == 0, K % 32 == 0.
__global__ void gemm_f16_wmma(const _Float16* __restrict__ A, const _Float16* __restrict__ Bp,
                              const float* __restrict__ bias, float* __restrict__ C,
                              int M, int N, int K, int lda,
                              long strideA, long strideB, long strideC) {
    __shared__ _Float16 lds[2][2048];   // 2 x 4KB
    int batch = blockIdx.y;
    A  += (long)batch * strideA;
    Bp += (long)batch * strideB;
    C  += (long)batch * strideC;

    const int lane = threadIdx.x & 31;
    const int wave = threadIdx.x >> 5;
    const int nStrips = N >> 6;
    const int mb = blockIdx.x / nStrips;
    const int ns = blockIdx.x % nStrips;
    const int Kt = K >> 5;

    // --- B staging assignment: thread -> (ntile tnt, 16B chunk tj)
    const int tnt = threadIdx.x >> 6;     // 0..3
    const int tj  = threadIdx.x & 63;     // 0..63
    const _Float16* bsrc = Bp + ((long)(ns * 4 + tnt) * Kt) * 512 + tj * 8;
    _Float16* dstBuf[2] = { &lds[0][tnt * 512 + tj * 8], &lds[1][tnt * 512 + tj * 8] };

    // --- A fragment addressing (per wave: one 16-row M tile)
    const int mt = mb * 8 + wave;
    const int m = mt * 16 + (lane & 15);
    const int koff = (lane & 16) ? 8 : 0;
    const _Float16* arow = A + (long)m * lda;

    v8f acc0 = {}, acc1 = {}, acc2 = {}, acc3 = {};

    // stage kt=0 into buffer 0
#if HAS_ASYNC_LDS
    __builtin_amdgcn_global_load_async_to_lds_b128(
        (GLOBAL_AS v4i*)bsrc, (LDS_AS v4i*)dstBuf[0], 0, 0);
#else
    *(v8h*)dstBuf[0] = *(const v8h*)bsrc;
#endif

    for (int kt = 0; kt < Kt; ++kt) {
        const int cur = kt & 1;
        // make staged data for this kt visible to the whole block
#if HAS_ASYNC_LDS
#if __has_builtin(__builtin_amdgcn_s_wait_asynccnt)
        __builtin_amdgcn_s_wait_asynccnt(0);
#else
        asm volatile("s_wait_asynccnt 0x0" ::: "memory");
#endif
#endif
        __syncthreads();

        // prefetch next K-step into the other buffer
        if (kt + 1 < Kt) {
#if HAS_ASYNC_LDS
            __builtin_amdgcn_global_load_async_to_lds_b128(
                (GLOBAL_AS v4i*)(bsrc + (long)(kt + 1) * 512),
                (LDS_AS v4i*)dstBuf[cur ^ 1], 0, 0);
#else
            *(v8h*)dstBuf[cur ^ 1] = *(const v8h*)(bsrc + (long)(kt + 1) * 512);
#endif
        }

        // A fragment: two 16B row loads
        const int kb = kt * 32;
        v8h alo = *(const v8h*)(arow + kb + koff);
        v8h ahi = *(const v8h*)(arow + kb + 16 + koff);
        v16h a;
        #pragma unroll
        for (int j = 0; j < 8; ++j) { a[j] = alo[j]; a[j + 8] = ahi[j]; }

        // B fragments from LDS (32B each)
        const _Float16* lb = &lds[cur][lane * 16];
        v16h b0 = *(const v16h*)(lb);
        v16h b1 = *(const v16h*)(lb + 512);
        v16h b2 = *(const v16h*)(lb + 1024);
        v16h b3 = *(const v16h*)(lb + 1536);

        acc0 = __builtin_amdgcn_wmma_f32_16x16x32_f16(false, a, false, b0, (short)0, acc0, false, false);
        acc1 = __builtin_amdgcn_wmma_f32_16x16x32_f16(false, a, false, b1, (short)0, acc1, false, false);
        acc2 = __builtin_amdgcn_wmma_f32_16x16x32_f16(false, a, false, b2, (short)0, acc2, false, false);
        acc3 = __builtin_amdgcn_wmma_f32_16x16x32_f16(false, a, false, b3, (short)0, acc3, false, false);
    }

    // epilogue: C/D fragment layout -> row-major C
    const int rowBase = mt * 16 + ((lane & 16) ? 8 : 0);
    const int col = lane & 15;
    #pragma unroll
    for (int t = 0; t < 4; ++t) {
        v8f acc = (t == 0) ? acc0 : (t == 1) ? acc1 : (t == 2) ? acc2 : acc3;
        int n = ns * 64 + t * 16 + col;
        float bv = bias ? bias[n] : 0.f;
        #pragma unroll
        for (int r = 0; r < 8; ++r)
            C[(long)(rowBase + r) * N + n] = acc[r] + bv;
    }
}

// ---------------------------------------------------------------- qs reduce: sigmoid-gate scalar per (b,l,h)
__global__ void qs_kernel(const float* __restrict__ qraw, const float* __restrict__ q_bias,
                          const float* __restrict__ mask, float* __restrict__ qs) {
    int idx = blockIdx.x * blockDim.x + threadIdx.x;   // (b*L+l)*H + h
    if (idx >= BL_ * H_) return;
    int h = idx % H_;
    int bl = idx / H_;
    const float* qr = qraw + (long)bl * DIM_ + h * HD_;
    const float* qb = q_bias + h * HD_;
    float sum = 0.f;
    #pragma unroll 8
    for (int d = 0; d < HD_; ++d) {
        float x = qr[d] - __expf(qb[d]);
        sum += 1.f / (1.f + __expf(-x));
    }
    qs[idx] = sum * (1.f / HD_) * mask[bl];
}

// ---------------------------------------------------------------- comb scale + repack to [BL,512] f16
__global__ void comb_scale_kernel(const float* __restrict__ comb_raw, const float* __restrict__ qs,
                                  _Float16* __restrict__ combF16) {
    long idx = (long)blockIdx.x * blockDim.x + threadIdx.x;   // ((b*H+h)*L + l)*64 + d
    if (idx >= (long)B_ * H_ * L_ * HD_) return;
    int d = (int)(idx & (HD_ - 1));
    long t = idx >> 6;
    int l = (int)(t % L_);
    int bh = (int)(t / L_);
    int h = bh % H_;
    int b = bh / H_;
    float sc = qs[((long)b * L_ + l) * H_ + h];
    combF16[((long)b * L_ + l) * DIM_ + h * HD_ + d] = (_Float16)(sc * comb_raw[idx]);
}

// ---------------------------------------------------------------- fused residual-add + LayerNorm
__global__ void add_ln_kernel(const float* __restrict__ a, const float* __restrict__ b,
                              const float* __restrict__ g, const float* __restrict__ beta,
                              float* __restrict__ outF32, _Float16* __restrict__ outF16) {
    int row = blockIdx.x;
    int tid = threadIdx.x;
    const float* ar = a + (long)row * DIM_;
    const float* br = b + (long)row * DIM_;
    __shared__ float s1[256], s2[256];
    float x0 = ar[tid] + br[tid];
    float x1 = ar[tid + 256] + br[tid + 256];
    s1[tid] = x0 + x1;
    s2[tid] = x0 * x0 + x1 * x1;
    __syncthreads();
    for (int off = 128; off > 0; off >>= 1) {
        if (tid < off) { s1[tid] += s1[tid + off]; s2[tid] += s2[tid + off]; }
        __syncthreads();
    }
    float mean = s1[0] * (1.f / DIM_);
    float var  = s2[0] * (1.f / DIM_) - mean * mean;
    float rstd = rsqrtf(var + 1e-5f);
    float y0 = (x0 - mean) * rstd * g[tid] + beta[tid];
    float y1 = (x1 - mean) * rstd * g[tid + 256] + beta[tid + 256];
    long base = (long)row * DIM_;
    if (outF32) { outF32[base + tid] = y0; outF32[base + tid + 256] = y1; }
    if (outF16) { outF16[base + tid] = (_Float16)y0; outF16[base + tid + 256] = (_Float16)y1; }
}

// ---------------------------------------------------------------- GELU (tanh approx) f32 -> f16
__global__ void gelu_kernel(const float* __restrict__ src, _Float16* __restrict__ dst, long n) {
    long i = (long)blockIdx.x * blockDim.x + threadIdx.x;
    if (i >= n) return;
    float x = src[i];
    float t = tanhf(0.7978845608028654f * (x + 0.044715f * x * x * x));
    dst[i] = (_Float16)(0.5f * x * (1.f + t));
}

// ---------------------------------------------------------------- host side
static inline int ceilDiv(long a, long b) { return (int)((a + b - 1) / b); }

extern "C" void kernel_launch(void* const* d_in, const int* in_sizes, int n_in,
                              void* d_out, int out_size, void* d_ws, size_t ws_size,
                              hipStream_t stream) {
    const float* tensor   = (const float*)d_in[0];
    const float* mask     = (const float*)d_in[1];
    const float* angle    = (const float*)d_in[2];
    const float* delta    = (const float*)d_in[3];
    const float* Qw       = (const float*)d_in[4];
    const float* q_bias   = (const float*)d_in[5];
    // d_in[6]=Kw, d_in[7]=Kb : dead code in reference
    const float* Vw       = (const float*)d_in[8];
    const float* Vb       = (const float*)d_in[9];
    const float* Ow       = (const float*)d_in[10];
    const float* Ob       = (const float*)d_in[11];
    const float* ln1_g    = (const float*)d_in[12];
    const float* ln1_b    = (const float*)d_in[13];
    const float* Win      = (const float*)d_in[14];
    const float* b_in     = (const float*)d_in[15];
    const float* Wout     = (const float*)d_in[16];
    const float* b_out    = (const float*)d_in[17];
    const float* ln2_g    = (const float*)d_in[18];
    const float* ln2_b    = (const float*)d_in[19];
    float* out = (float*)d_out;

    // ---- workspace carve
    char* w = (char*)d_ws;
    auto carve = [&](size_t bytes) -> void* {
        void* p = (void*)w;
        w += (bytes + 255) & ~(size_t)255;
        return p;
    };
    _Float16* tensorF16 = (_Float16*)carve((size_t)BL_ * DIM_ * 2);
    float*    buf0      = (float*)   carve((size_t)BL_ * DIM_ * 4);   // qraw -> attn_out
    float*    buf1      = (float*)   carve((size_t)BL_ * DIM_ * 4);   // v -> lin
    float*    qs        = (float*)   carve((size_t)BL_ * H_ * 4);
    _Float16* vpack     = (_Float16*)carve((size_t)B_ * H_ * L_ * HD_ * 2);
    _Float16* trm       = (_Float16*)carve((size_t)B_ * H_ * L_ * TD2_ * 2);  // [bh][L][128]
    _Float16* ttr       = (_Float16*)carve((size_t)B_ * H_ * TD2_ * L_ * 2);  // [bh][128][L]
    float*    Wf32      = (float*)   carve((size_t)B_ * H_ * TD2_ * HD_ * 4);
    _Float16* Wpack     = (_Float16*)carve((size_t)B_ * H_ * TD2_ * HD_ * 2);
    float*    comb_raw  = (float*)   carve((size_t)B_ * H_ * L_ * HD_ * 4);
    _Float16* combF16   = (_Float16*)carve((size_t)BL_ * DIM_ * 2);
    float*    interF32  = (float*)   carve((size_t)BL_ * DIM_ * 4);
    _Float16* interF16  = (_Float16*)carve((size_t)BL_ * DIM_ * 2);
    float*    hid       = (float*)   carve((size_t)BL_ * LIN_ * 4);
    _Float16* hidF16    = (_Float16*)carve((size_t)BL_ * LIN_ * 2);
    _Float16* QwP       = (_Float16*)carve((size_t)DIM_ * DIM_ * 2);
    _Float16* VwP       = (_Float16*)carve((size_t)DIM_ * DIM_ * 2);
    _Float16* OwP       = (_Float16*)carve((size_t)DIM_ * DIM_ * 2);
    _Float16* WinP      = (_Float16*)carve((size_t)DIM_ * LIN_ * 2);
    _Float16* WoutP     = (_Float16*)carve((size_t)LIN_ * DIM_ * 2);
    (void)ws_size; (void)in_sizes; (void)n_in; (void)out_size;

    // ---- stage 0: conversions / packing / time features
    long nTen = (long)BL_ * DIM_;
    cvt_f32_f16<<<ceilDiv(nTen, 256), 256, 0, stream>>>(tensor, tensorF16, nTen);
    time_kernel<<<ceilDiv((long)H_ * L_ * TD2_, 256), 256, 0, stream>>>(angle, delta, trm, ttr);
    pack_b_f32<<<dim3(ceilDiv((DIM_/16)*(DIM_/32)*32, 256), 1), 256, 0, stream>>>(Qw,  QwP,  DIM_, DIM_, DIM_, 0, 0);
    pack_b_f32<<<dim3(ceilDiv((DIM_/16)*(DIM_/32)*32, 256), 1), 256, 0, stream>>>(Vw,  VwP,  DIM_, DIM_, DIM_, 0, 0);
    pack_b_f32<<<dim3(ceilDiv((DIM_/16)*(DIM_/32)*32, 256), 1), 256, 0, stream>>>(Ow,  OwP,  DIM_, DIM_, DIM_, 0, 0);
    pack_b_f32<<<dim3(ceilDiv((LIN_/16)*(DIM_/32)*32, 256), 1), 256, 0, stream>>>(Win, WinP, DIM_, LIN_, LIN_, 0, 0);
    pack_b_f32<<<dim3(ceilDiv((DIM_/16)*(LIN_/32)*32, 256), 1), 256, 0, stream>>>(Wout,WoutP, LIN_, DIM_, DIM_, 0, 0);

    // ---- q / v projections : [4096,512] x [512,512]
    {
        int blocks = (BL_ / 128) * (DIM_ / 64);  // 256
        gemm_f16_wmma<<<dim3(blocks, 1), 256, 0, stream>>>(
            tensorF16, QwP, nullptr, buf0, BL_, DIM_, DIM_, DIM_, 0, 0, 0);
        gemm_f16_wmma<<<dim3(blocks, 1), 256, 0, stream>>>(
            tensorF16, VwP, Vb, buf1, BL_, DIM_, DIM_, DIM_, 0, 0, 0);
    }

    qs_kernel<<<ceilDiv((long)BL_ * H_, 256), 256, 0, stream>>>(buf0, q_bias, mask, qs);

    // pack v[b,h] (K=L rows, N=64 cols) into fragment layout (Vb already folded via GEMM bias)
    for (int b = 0; b < B_; ++b) {
        pack_b_f32<<<dim3(ceilDiv((HD_/16)*(L_/32)*32, 256), H_), 256, 0, stream>>>(
            buf1 + (long)b * L_ * DIM_, vpack + (long)b * H_ * L_ * HD_,
            L_, HD_, DIM_, /*srcBatchStride=*/HD_, /*dstBatchStride=*/(long)L_ * HD_);
    }

    // ---- W[b,h] = time_h^T [128,2048] @ v[b,h] [2048,64]  (16 batches)
    gemm_f16_wmma<<<dim3((TD2_/128)*(HD_/64), B_ * H_), 256, 0, stream>>>(
        ttr, vpack, nullptr, Wf32, TD2_, HD_, L_, L_,
        (long)TD2_ * L_, (long)L_ * HD_, (long)TD2_ * HD_);
    pack_b_f32<<<dim3(ceilDiv((HD_/16)*(TD2_/32)*32, 256), B_ * H_), 256, 0, stream>>>(
        Wf32, Wpack, TD2_, HD_, HD_, (long)TD2_ * HD_, (long)TD2_ * HD_);

    // ---- comb_raw[b,h] = time[b,h] [2048,128] @ W[b,h] [128,64]
    gemm_f16_wmma<<<dim3((L_/128)*(HD_/64), B_ * H_), 256, 0, stream>>>(
        trm, Wpack, nullptr, comb_raw, L_, HD_, TD2_, TD2_,
        (long)L_ * TD2_, (long)TD2_ * HD_, (long)L_ * HD_);
    comb_scale_kernel<<<ceilDiv((long)B_ * H_ * L_ * HD_, 256), 256, 0, stream>>>(comb_raw, qs, combF16);

    // ---- attn_out = comb @ Ow + Ob ; inter = LN(tensor + attn_out)
    gemm_f16_wmma<<<dim3((BL_/128)*(DIM_/64), 1), 256, 0, stream>>>(
        combF16, OwP, Ob, buf0, BL_, DIM_, DIM_, DIM_, 0, 0, 0);
    add_ln_kernel<<<BL_, 256, 0, stream>>>(tensor, buf0, ln1_g, ln1_b, interF32, interF16);

    // ---- FFN: hid = gelu(inter @ Win + b_in) ; lin = hid @ Wout + b_out
    gemm_f16_wmma<<<dim3((BL_/128)*(LIN_/64), 1), 256, 0, stream>>>(
        interF16, WinP, b_in, hid, BL_, LIN_, DIM_, DIM_, 0, 0, 0);
    gelu_kernel<<<ceilDiv((long)BL_ * LIN_, 256), 256, 0, stream>>>(hid, hidF16, (long)BL_ * LIN_);
    gemm_f16_wmma<<<dim3((BL_/128)*(DIM_/64), 1), 256, 0, stream>>>(
        hidF16, WoutP, b_out, buf1, BL_, DIM_, LIN_, LIN_, 0, 0, 0);

    // ---- out = LN(inter + lin)
    add_ln_kernel<<<BL_, 256, 0, stream>>>(interF32, buf1, ln2_g, ln2_b, out, nullptr);
}
